// GNNAttention_39410619908366
// MI455X (gfx1250) — matrix operand order repaired and checked by reading, not compiled, not run on hardware
//
#include <hip/hip_runtime.h>
#include <hip/hip_bf16.h>
#include <climits>

typedef __attribute__((ext_vector_type(16))) _Float16 v16h;
typedef __attribute__((ext_vector_type(8)))  _Float16 h8;
typedef __attribute__((ext_vector_type(8)))  float    v8f;

// ---------- helpers ----------
__device__ __forceinline__ int f2o(float f) {           // monotone float -> int
    int i = __float_as_int(f);
    return (i >= 0) ? i : (i ^ 0x7fffffff);
}
__device__ __forceinline__ float o2f(int i) {
    return __int_as_float((i >= 0) ? i : (i ^ 0x7fffffff));
}

// ---------- generic fills / converts ----------
__global__ void k_fill_f32(float* p, float v, size_t n) {
    size_t i = (size_t)blockIdx.x * blockDim.x + threadIdx.x;
    size_t st = (size_t)gridDim.x * blockDim.x;
    for (; i < n; i += st) p[i] = v;
}
__global__ void k_fill_i32(int* p, int v, size_t n) {
    size_t i = (size_t)blockIdx.x * blockDim.x + threadIdx.x;
    size_t st = (size_t)gridDim.x * blockDim.x;
    for (; i < n; i += st) p[i] = v;
}
__global__ void k_f32_to_f16(const float* __restrict__ in, _Float16* __restrict__ out, size_t n) {
    size_t i = (size_t)blockIdx.x * blockDim.x + threadIdx.x;
    size_t st = (size_t)gridDim.x * blockDim.x;
    for (; i < n; i += st) out[i] = (_Float16)in[i];
}

// ---------- GraphConv neighbor scatter-add: nbr[dst] += x[src], F = 128 ----------
__global__ void k_scatter_add128(const float* __restrict__ x, const int* __restrict__ src,
                                 const int* __restrict__ dst, float* __restrict__ nbr,
                                 long long E) {
    size_t total = (size_t)E << 7;                       // E * 128
    size_t i = (size_t)blockIdx.x * blockDim.x + threadIdx.x;
    size_t st = (size_t)gridDim.x * blockDim.x;
    for (; i < total; i += st) {
        size_t e = i >> 7;
        int f = (int)(i & 127);
        atomicAdd(&nbr[(size_t)dst[e] * 128 + f], x[(size_t)src[e] * 128 + f]);
    }
}

// ---------- WMMA GEMM: C[M x Nout] = A[M x K] * Bw^T (+ A2 * Bw2^T) (+bias) ----------
// Bw is [Nout x K] row-major f16 (so B[k][n] = Bw[n*K+k], loaded column-wise per lane).
// One wave computes one 16x16 tile via v_wmma_f32_16x16x32_f16, K stepped by 32.
__device__ __forceinline__ v8f wmma_k_loop(const _Float16* __restrict__ aRow,
                                           const _Float16* __restrict__ bRow,
                                           int K, int hh, v8f acc) {
    union F16x16 { v16h v; h8 h[2]; };
    for (int kb = 0; kb < K; kb += 32) {
        F16x16 a, b;
        // A frag: lane holds row (lane&15); halves 0..7 = K kb+8*hh.., 8..15 = K kb+16+8*hh..
        a.h[0] = *(const h8*)(aRow + kb + 8 * hh);
        a.h[1] = *(const h8*)(aRow + kb + 16 + 8 * hh);
        // B frag: lane holds column (lane&15); halves 0..7 = K kb+8*hh.., 8..15 = K kb+16+8*hh..
        b.h[0] = *(const h8*)(bRow + kb + 8 * hh);
        b.h[1] = *(const h8*)(bRow + kb + 16 + 8 * hh);
        acc = __builtin_amdgcn_wmma_f32_16x16x32_f16(false, a.v, false, b.v,
                                                     (short)0, acc, false, false);
    }
    return acc;
}

__global__ __launch_bounds__(256)
void k_gemm_wmma(const _Float16* __restrict__ A, const _Float16* __restrict__ Bw,
                 const _Float16* __restrict__ A2, const _Float16* __restrict__ Bw2,
                 const float* __restrict__ bias,
                 _Float16* __restrict__ outh, float* __restrict__ outf,
                 int M, int Nout, int K, int do_elu) {
    int gtid = blockIdx.x * blockDim.x + threadIdx.x;
    int wave = gtid >> 5;
    int lane = gtid & 31;
    int tilesN = Nout >> 4;
    int tiles  = (M >> 4) * tilesN;
    if (wave >= tiles) return;
    int m0 = (wave / tilesN) << 4;
    int n0 = (wave % tilesN) << 4;
    int r  = lane & 15;
    int hh = lane >> 4;

    v8f acc = {};
    acc = wmma_k_loop(A + (size_t)(m0 + r) * K, Bw + (size_t)(n0 + r) * K, K, hh, acc);
    if (A2)
        acc = wmma_k_loop(A2 + (size_t)(m0 + r) * K, Bw2 + (size_t)(n0 + r) * K, K, hh, acc);

    int nn = n0 + r;
    float bv = bias ? bias[nn] : 0.f;
#pragma unroll
    for (int v = 0; v < 8; v++) {                        // C/D: M = v + 8*(lane>>4), N = lane&15
        int mm = m0 + v + 8 * hh;
        float cv = acc[v] + bv;
        if (do_elu) cv = cv > 0.f ? cv : (__expf(cv) - 1.f);
        size_t o = (size_t)mm * Nout + nn;
        if (outf) outf[o] = cv;
        if (outh) outh[o] = (_Float16)cv;
    }
}

// ---------- GAT per-node attention coefficients ----------
__global__ void k_alpha(const float* __restrict__ z, const float* __restrict__ a_s,
                        const float* __restrict__ a_d, float* __restrict__ as,
                        float* __restrict__ ad, int Nn, int H, int C) {
    long long idx = (long long)blockIdx.x * blockDim.x + threadIdx.x;
    if (idx >= (long long)Nn * H) return;
    int n = (int)(idx / H), hh = (int)(idx % H);
    const float* zp  = z + (size_t)n * H * C + (size_t)hh * C;
    const float* asp = a_s + hh * C;
    const float* adp = a_d + hh * C;
    float ss = 0.f, sd = 0.f;
    for (int c = 0; c < C; c++) { float zc = zp[c]; ss += zc * asp[c]; sd += zc * adp[c]; }
    as[idx] = ss; ad[idx] = sd;
}

// ---------- edge softmax pass 1: segment max (self-loops appended) ----------
__global__ void k_edge_max(const int* __restrict__ src, const int* __restrict__ dst,
                           const float* __restrict__ as, const float* __restrict__ ad,
                           int* __restrict__ m, long long E, int Nn, int H) {
    long long idx = (long long)blockIdx.x * blockDim.x + threadIdx.x;
    long long tot = (E + Nn) * (long long)H;
    if (idx >= tot) return;
    long long e = idx / H; int hh = (int)(idx % H);
    int s = (e < E) ? src[e] : (int)(e - E);
    int d = (e < E) ? dst[e] : (int)(e - E);
    float l = as[(size_t)s * H + hh] + ad[(size_t)d * H + hh];
    l = l > 0.f ? l : 0.2f * l;                          // leaky_relu(0.2)
    atomicMax(&m[(size_t)d * H + hh], f2o(l));
}

// ---------- edge softmax pass 2: unnormalized exp-weighted aggregation ----------
__global__ void k_edge_accum(const int* __restrict__ src, const int* __restrict__ dst,
                             const float* __restrict__ as, const float* __restrict__ ad,
                             const int* __restrict__ m, const float* __restrict__ z,
                             float* __restrict__ agg, float* __restrict__ denom,
                             long long E, int Nn, int H, int C) {
    long long idx = (long long)blockIdx.x * blockDim.x + threadIdx.x;
    int HC = H * C;
    long long tot = (E + Nn) * (long long)HC;
    if (idx >= tot) return;
    long long e = idx / HC; int rem = (int)(idx % HC);
    int hh = rem / C; int c = rem % C;
    int s = (e < E) ? src[e] : (int)(e - E);
    int d = (e < E) ? dst[e] : (int)(e - E);
    float l = as[(size_t)s * H + hh] + ad[(size_t)d * H + hh];
    l = l > 0.f ? l : 0.2f * l;
    float ex = __expf(l - o2f(m[(size_t)d * H + hh]));
    atomicAdd(&agg[(size_t)d * HC + rem], ex * z[(size_t)s * HC + rem]);
    if (c == 0) atomicAdd(&denom[(size_t)d * H + hh], ex);
}

// ---------- GAT epilogue: normalize, +bias, optional ELU; f16 and/or f32 out ----------
__global__ void k_finalize(const float* __restrict__ agg, const float* __restrict__ denom,
                           const float* __restrict__ bias, _Float16* __restrict__ outh,
                           float* __restrict__ outf, int Nn, int H, int C, int do_elu) {
    long long idx = (long long)blockIdx.x * blockDim.x + threadIdx.x;
    int HC = H * C;
    if (idx >= (long long)Nn * HC) return;
    int n = (int)(idx / HC); int rem = (int)(idx % HC); int hh = rem / C;
    float v = agg[idx] / (denom[(size_t)n * H + hh] + 1e-16f) + bias[rem];
    if (do_elu) v = v > 0.f ? v : (__expf(v) - 1.f);
    if (outh) outh[idx] = (_Float16)v;
    if (outf) outf[idx] = v;
}

// ---------- global mean + final linear ----------
__global__ __launch_bounds__(256)
void k_colsum(const float* __restrict__ x, float* __restrict__ gsum, int Nn, int C) {
    int c = blockIdx.x;                                  // one block per channel
    float s = 0.f;
    for (int n = threadIdx.x; n < Nn; n += blockDim.x) s += x[(size_t)n * C + c];
    __shared__ float sm[256];
    sm[threadIdx.x] = s; __syncthreads();
    for (int off = 128; off > 0; off >>= 1) {
        if ((int)threadIdx.x < off) sm[threadIdx.x] += sm[threadIdx.x + off];
        __syncthreads();
    }
    if (threadIdx.x == 0) gsum[c] = sm[0];
}
__global__ void k_final(const float* __restrict__ gsum, const float* __restrict__ Wr,
                        const float* __restrict__ br, float* __restrict__ out, int Nn, int C) {
    __shared__ float sm[64];
    int c = threadIdx.x;
    sm[c] = (gsum[c] / (float)Nn) * Wr[c];
    __syncthreads();
    if (c == 0) {
        float t = 0.f;
        for (int i = 0; i < C; i++) t += sm[i];
        out[0] = t + br[0];
    }
}

// ====================================================================
extern "C" void kernel_launch(void* const* d_in, const int* in_sizes, int n_in,
                              void* d_out, int out_size, void* d_ws, size_t ws_size,
                              hipStream_t stream) {
    const int       Nn = 50000;
    const long long Ee = 800000;
    const int       F  = 128;

    const float* x      = (const float*)d_in[0];
    const int*   ei     = (const int*)d_in[1];
    const int*   srcv   = ei;
    const int*   dstv   = ei + Ee;
    const float* W_rel  = (const float*)d_in[2];
    const float* b_rel  = (const float*)d_in[3];
    const float* W_root = (const float*)d_in[4];
    const float* W1     = (const float*)d_in[5];
    const float* a1s    = (const float*)d_in[6];
    const float* a1d    = (const float*)d_in[7];
    const float* b1     = (const float*)d_in[8];
    const float* W2     = (const float*)d_in[9];
    const float* a2s    = (const float*)d_in[10];
    const float* a2d    = (const float*)d_in[11];
    const float* b2     = (const float*)d_in[12];
    const float* Wr     = (const float*)d_in[13];
    const float* br     = (const float*)d_in[14];
    float*       outp   = (float*)d_out;

    // ---- workspace layout (lifetime-based aliasing, peak ~78 MB) ----
    char* ws = (char*)d_ws;
    size_t off = 0;
    auto alloc = [&](size_t bytes) -> char* {
        char* p = ws + off; off += (bytes + 255) & ~(size_t)255; return p;
    };
    _Float16* xh     = (_Float16*)alloc((size_t)Nn * F * 2);   // f16 x      (dead after GEMM1)
    _Float16* nbrh   = (_Float16*)alloc((size_t)Nn * F * 2);   // f16 nbr    (dead after GEMM1)
    _Float16* h1     = (_Float16*)alloc((size_t)Nn * F * 2);   // GraphConv out (dead after z1)
    float*    nbr    = (float*)   alloc((size_t)Nn * F * 4);   // f32 nbr    (dead after convert)
    _Float16* h2     = (_Float16*)alloc((size_t)Nn * 64 * 2);  // GAT1 out f16
    float*    as1    = (float*)   alloc((size_t)Nn * 8 * 4);
    float*    ad1    = (float*)   alloc((size_t)Nn * 8 * 4);
    int*      m1     = (int*)     alloc((size_t)Nn * 8 * 4);
    float*    den1   = (float*)   alloc((size_t)Nn * 8 * 4);
    float*    as2    = (float*)   alloc((size_t)Nn * 4);
    float*    ad2    = (float*)   alloc((size_t)Nn * 4);
    int*      m2     = (int*)     alloc((size_t)Nn * 4);
    float*    den2   = (float*)   alloc((size_t)Nn * 4);
    float*    gsum   = (float*)   alloc(64 * 4);
    _Float16* wrelh  = (_Float16*)alloc(128 * 128 * 2);
    _Float16* wrooth = (_Float16*)alloc(128 * 128 * 2);
    _Float16* w1h    = (_Float16*)alloc(64 * 128 * 2);
    _Float16* w2h    = (_Float16*)alloc(64 * 64 * 2);
    // aliases (strictly later lifetimes):
    float* z1   = nbr;                         // N*64 f32, after nbr is consumed
    float* agg1 = nbr + (size_t)Nn * 64;       // N*64 f32
    float* z2   = (float*)xh;                  // N*64 f32
    float* agg2 = (float*)nbrh;                // N*64 f32
    float* out2 = (float*)h1;                  // N*64 f32

    auto gs = [](size_t n) { return dim3((unsigned)((n + 255) / 256)); };

    // ---- precision conversions ----
    k_f32_to_f16<<<gs((size_t)Nn * F), 256, 0, stream>>>(x, xh, (size_t)Nn * F);
    k_f32_to_f16<<<gs(128 * 128), 256, 0, stream>>>(W_rel, wrelh, 128 * 128);
    k_f32_to_f16<<<gs(128 * 128), 256, 0, stream>>>(W_root, wrooth, 128 * 128);
    k_f32_to_f16<<<gs(64 * 128), 256, 0, stream>>>(W1, w1h, 64 * 128);
    k_f32_to_f16<<<gs(64 * 64), 256, 0, stream>>>(W2, w2h, 64 * 64);

    // ---- GraphConv: nbr = segment_sum(x[src], dst) ----
    k_fill_f32<<<gs((size_t)Nn * F), 256, 0, stream>>>(nbr, 0.f, (size_t)Nn * F);
    k_scatter_add128<<<gs((size_t)Ee * F), 256, 0, stream>>>(x, srcv, dstv, nbr, Ee);
    k_f32_to_f16<<<gs((size_t)Nn * F), 256, 0, stream>>>(nbr, nbrh, (size_t)Nn * F);

    // h1 = ELU(nbr @ W_rel^T + x @ W_root^T + b_rel)   [N x 128], f16 out
    {
        size_t waves = (size_t)(Nn / 16) * (F / 16);
        k_gemm_wmma<<<gs(waves * 32), 256, 0, stream>>>(nbrh, wrelh, xh, wrooth, b_rel,
                                                        h1, nullptr, Nn, F, F, 1);
    }

    // ---- GAT layer 1 (H=8, C=8) ----
    {
        size_t waves = (size_t)(Nn / 16) * (64 / 16);
        k_gemm_wmma<<<gs(waves * 32), 256, 0, stream>>>(h1, w1h, nullptr, nullptr, nullptr,
                                                        nullptr, z1, Nn, 64, 128, 0);
    }
    k_alpha<<<gs((size_t)Nn * 8), 256, 0, stream>>>(z1, a1s, a1d, as1, ad1, Nn, 8, 8);
    k_fill_i32<<<gs((size_t)Nn * 8), 256, 0, stream>>>(m1, INT_MIN, (size_t)Nn * 8);
    k_fill_f32<<<gs((size_t)Nn * 8), 256, 0, stream>>>(den1, 0.f, (size_t)Nn * 8);
    k_fill_f32<<<gs((size_t)Nn * 64), 256, 0, stream>>>(agg1, 0.f, (size_t)Nn * 64);
    k_edge_max<<<gs((size_t)(Ee + Nn) * 8), 256, 0, stream>>>(srcv, dstv, as1, ad1, m1, Ee, Nn, 8);
    k_edge_accum<<<gs((size_t)(Ee + Nn) * 64), 256, 0, stream>>>(srcv, dstv, as1, ad1, m1, z1,
                                                                 agg1, den1, Ee, Nn, 8, 8);
    k_finalize<<<gs((size_t)Nn * 64), 256, 0, stream>>>(agg1, den1, b1, h2, nullptr, Nn, 8, 8, 1);

    // ---- GAT layer 2 (H=1, C=64) ----
    {
        size_t waves = (size_t)(Nn / 16) * (64 / 16);
        k_gemm_wmma<<<gs(waves * 32), 256, 0, stream>>>(h2, w2h, nullptr, nullptr, nullptr,
                                                        nullptr, z2, Nn, 64, 64, 0);
    }
    k_alpha<<<gs((size_t)Nn), 256, 0, stream>>>(z2, a2s, a2d, as2, ad2, Nn, 1, 64);
    k_fill_i32<<<gs((size_t)Nn), 256, 0, stream>>>(m2, INT_MIN, (size_t)Nn);
    k_fill_f32<<<gs((size_t)Nn), 256, 0, stream>>>(den2, 0.f, (size_t)Nn);
    k_fill_f32<<<gs((size_t)Nn * 64), 256, 0, stream>>>(agg2, 0.f, (size_t)Nn * 64);
    k_edge_max<<<gs((size_t)(Ee + Nn)), 256, 0, stream>>>(srcv, dstv, as2, ad2, m2, Ee, Nn, 1);
    k_edge_accum<<<gs((size_t)(Ee + Nn) * 64), 256, 0, stream>>>(srcv, dstv, as2, ad2, m2, z2,
                                                                 agg2, den2, Ee, Nn, 1, 64);
    k_finalize<<<gs((size_t)Nn * 64), 256, 0, stream>>>(agg2, den2, b2, nullptr, out2, Nn, 1, 64, 0);

    // ---- global mean + regression head ----
    k_colsum<<<dim3(64), 256, 0, stream>>>(out2, gsum, Nn, 64);
    k_final<<<dim3(1), 64, 0, stream>>>(gsum, Wr, br, outp, Nn, 64);
}
